// MultiHeadAttention_20753281974875
// MI455X (gfx1250) — compile-verified, matrix operand order checked by reading, not run on hardware
//
#include <hip/hip_runtime.h>
#include <hip/hip_bf16.h>
#include <cstdint>
#include <cstddef>

// ============================================================================
// Multi-head attention forward (B=4, S=2048, d_model=1024, 16 heads, d_k=64)
// bf16 WMMA (v_wmma_f32_16x16x32_bf16) for all six GEMMs, f32 accumulate.
// attns (1.07 GB f32) dominates HBM traffic -> bandwidth bound.
// Direct async global->LDS staging (GLOBAL_LOAD_ASYNC_TO_LDS_B128) for all
// straight-copy tiles, plus global prefetch of the next K-slice.
// ============================================================================

static constexpr int NH = 16;          // heads
static constexpr int DM = 1024;        // d_model
static constexpr int DK = 64;          // d_k == d_v
static constexpr int BS = 4;           // batch
static constexpr int SQ = 2048;        // sequence length
static constexpr int ROWS = BS * SQ;   // 8192 token rows
static constexpr float INV_TEMPER = 0.03125f;   // 1/sqrt(1024)
static constexpr float LN_EPS = 1e-5f;

// ---------------- workspace layout (bytes), total ~159.4 MB ----------------
static constexpr size_t SZ_XBF = (size_t)ROWS * DM * 2;        // 16 MiB
static constexpr size_t SZ_WBF = (size_t)NH * DM * DK * 2;     //  2 MiB
static constexpr size_t SZ_HBF = (size_t)NH * ROWS * DK * 2;   // 16 MiB

static constexpr size_t OFF_QBF = 0;                    // q  as bf16 [8192,1024]
static constexpr size_t OFF_KBF = OFF_QBF + SZ_XBF;     // k  as bf16
static constexpr size_t OFF_VBF = OFF_KBF + SZ_XBF;     // v  as bf16
static constexpr size_t OFF_WQ  = OFF_VBF + SZ_XBF;     // w_qs bf16 [16,1024,64]
static constexpr size_t OFF_WK  = OFF_WQ  + SZ_WBF;
static constexpr size_t OFF_WV  = OFF_WK  + SZ_WBF;
static constexpr size_t OFF_PW  = OFF_WV  + SZ_WBF;     // proj_w bf16 [1024,1024]
static constexpr size_t OFF_QH  = OFF_PW  + (size_t)DM * DM * 2; // qh bf16 [16,8192,64]
static constexpr size_t OFF_KH  = OFF_QH  + SZ_HBF;
static constexpr size_t OFF_VH  = OFF_KH  + SZ_HBF;
static constexpr size_t OFF_OH  = OFF_VH  + SZ_HBF;     // concat heads bf16 [8192,1024]
static constexpr size_t OFF_Y   = OFF_OH  + SZ_XBF;     // pre-LN f32 [8192,1024]

// ---------------- WMMA fragment helpers ----------------
typedef __attribute__((ext_vector_type(16))) __bf16 v16bf;
typedef __attribute__((ext_vector_type(8)))  float  v8f;

union FragBF {
  v16bf v;
  unsigned int w[8];
};

__device__ __forceinline__ unsigned short f2bf(float f) {
  unsigned int x = __float_as_uint(f);
  unsigned int r = x + 0x7FFFu + ((x >> 16) & 1u);   // round-to-nearest-even
  return (unsigned short)(r >> 16);
}

// 16-bit A-matrix 16x32 VGPR layout (ISA 7.12.2): pair p of a lane holds
// K = {0,2,4,6,16,18,20,22}[p] + 8*(lane>=16), plus its K+1 neighbor.
__device__ __forceinline__ int frag_k(int p, int lane) {
  int k = (p < 4) ? (2 * p) : (16 + 2 * (p - 4));
  return k + ((lane & 16) >> 1);
}

static constexpr int LDK = 40;   // LDS K-stride (32 + pad), keeps 16B alignment

// ---------------- async global->LDS staging (CDNA5 path) ----------------
#if defined(__HIP_DEVICE_COMPILE__) && defined(__gfx1250__) && \
    __has_builtin(__builtin_amdgcn_global_load_async_to_lds_b128)
#define USE_ASYNC_LDS 1
#else
#define USE_ASYNC_LDS 0
#endif

#if USE_ASYNC_LDS
typedef __attribute__((ext_vector_type(4))) int v4i_;
typedef __attribute__((address_space(1))) v4i_ gv4i;   // global 4xi32
typedef __attribute__((address_space(3))) v4i_ lv4i;   // LDS 4xi32
#endif

// 16-byte straight copy global -> LDS (per lane). Async path uses
// GLOBAL_LOAD_ASYNC_TO_LDS_B128 (tracked by ASYNCcnt, no VGPR round-trip).
__device__ __forceinline__ void stage16(const unsigned short* g, unsigned short* l) {
#if USE_ASYNC_LDS
  __builtin_amdgcn_global_load_async_to_lds_b128((gv4i*)g, (lv4i*)l, 0, 0);
#else
  *(uint4*)l = *(const uint4*)g;
#endif
}

// Drain this wave's ASYNCcnt before the workgroup barrier.
__device__ __forceinline__ void stage_fence() {
#if USE_ASYNC_LDS
  asm volatile("s_wait_asynccnt 0x0" ::: "memory");
#endif
}

// Build A (m = rBase+lane) and two B fragments (n = cBase+lane, +16) from LDS
// tiles stored [row][k] with K contiguous, and accumulate one K=32 step.
__device__ __forceinline__ void wmma_step(const unsigned short* Asm,
                                          const unsigned short* Bsm,
                                          int rBase, int cBase, int lane,
                                          v8f& acc0, v8f& acc1) {
  FragBF fa, fb0, fb1;
  const int m = rBase + (lane & 15);
  const int n = cBase + (lane & 15);
#pragma unroll
  for (int p = 0; p < 8; ++p) {
    const int kf = frag_k(p, lane);
    fa.w[p]  = *(const unsigned int*)&Asm[m * LDK + kf];
    fb0.w[p] = *(const unsigned int*)&Bsm[n * LDK + kf];
    fb1.w[p] = *(const unsigned int*)&Bsm[(n + 16) * LDK + kf];
  }
  acc0 = __builtin_amdgcn_wmma_f32_16x16x32_bf16(false, fa.v, false, fb0.v,
                                                 (short)0, acc0, false, false);
  acc1 = __builtin_amdgcn_wmma_f32_16x16x32_bf16(false, fa.v, false, fb1.v,
                                                 (short)0, acc1, false, false);
}

// ---------------- kernel 0: f32 -> bf16 convert ----------------
__global__ void __launch_bounds__(256)
cvt_f32_bf16(const float* __restrict__ s, unsigned short* __restrict__ d, int n) {
  int i = blockIdx.x * 256 + threadIdx.x;
  if (i < n) d[i] = f2bf(s[i]);
}

// ---------------- kernel 1: per-head input projections ----------------
// grid.z = tensor(0..2)*16 + head; block computes 64 rows x 64 cols (full d_k)
__global__ void __launch_bounds__(256)
gemm_head_proj(char* ws) {
  __shared__ __align__(16) unsigned short Asm[64 * LDK];
  __shared__ __align__(16) unsigned short Bsm[64 * LDK];   // [n][k] transposed

  const int z = blockIdx.z;
  const int t = z >> 4;          // 0=q 1=k 2=v
  const int h = z & 15;
  const unsigned short* X =
      (const unsigned short*)(ws + (t == 0 ? OFF_QBF : t == 1 ? OFF_KBF : OFF_VBF));
  const unsigned short* W =
      (const unsigned short*)(ws + (t == 0 ? OFF_WQ : t == 1 ? OFF_WK : OFF_WV)) +
      (size_t)h * DM * DK;
  unsigned short* Out =
      (unsigned short*)(ws + (t == 0 ? OFF_QH : t == 1 ? OFF_KH : OFF_VH)) +
      (size_t)h * ROWS * DK;

  const int rowBase = blockIdx.x * 64;
  const int tid  = threadIdx.x;
  const int lane = tid & 31;
  const int wv   = tid >> 5;            // 8 wave32s
  const int rB   = (wv & 3) * 16;
  const int cB   = (wv >> 2) * 32;

  v8f acc0 = {}; v8f acc1 = {};

  for (int k0 = 0; k0 < DM; k0 += 32) {
    {  // A tile: 64 rows x 32 k, straight async copy (128-bit per lane)
      const int idx = tid * 8, r = idx >> 5, kk = idx & 31;
      const unsigned short* src = X + (size_t)(rowBase + r) * DM + k0 + kk;
      stage16(src, &Asm[r * LDK + kk]);
      if (k0 + 32 < DM) __builtin_prefetch(src + 32, 0, 1);
    }
    {  // W tile: 32 k x 64 n, transpose into Bsm[n][k]
      const int idx = tid * 8, kk = idx >> 6, n0 = idx & 63;
      uint4 d = *(const uint4*)(W + (size_t)(k0 + kk) * DK + n0);
      const unsigned short* u = (const unsigned short*)&d;
#pragma unroll
      for (int i = 0; i < 8; ++i) Bsm[(n0 + i) * LDK + kk] = u[i];
    }
    stage_fence();
    __syncthreads();
    wmma_step(Asm, Bsm, rB, cB, lane, acc0, acc1);
    __syncthreads();
  }

  const int n = cB + (lane & 15);
#pragma unroll
  for (int r8 = 0; r8 < 8; ++r8) {
    const int m = rB + r8 + ((lane & 16) >> 1);
    const size_t row = (size_t)rowBase + m;
    Out[row * DK + n]      = f2bf(acc0[r8]);
    Out[row * DK + n + 16] = f2bf(acc1[r8]);
  }
}

// ---------------- kernel 2a: raw attention scores ----------------
// grid = (q-tile 32, key-tile 32, hb 64); scores[q][s] = (qh . kh)/temper
__global__ void __launch_bounds__(256)
attn_scores(const unsigned short* __restrict__ qh,
            const unsigned short* __restrict__ kh,
            float* __restrict__ attns) {
  __shared__ __align__(16) unsigned short Qs[64 * LDK];
  __shared__ __align__(16) unsigned short Ks[64 * LDK];

  const int hb = blockIdx.z;                         // h*4 + b
  const unsigned short* Q = qh + (size_t)hb * SQ * DK;  // h*8192+b*2048 == hb*2048
  const unsigned short* K = kh + (size_t)hb * SQ * DK;
  const int qBase = blockIdx.x * 64;
  const int sBase = blockIdx.y * 64;

  const int tid = threadIdx.x, lane = tid & 31, wv = tid >> 5;
  const int rB = (wv & 3) * 16, cB = (wv >> 2) * 32;

  v8f acc0 = {}; v8f acc1 = {};

#pragma unroll
  for (int k0 = 0; k0 < DK; k0 += 32) {
    const int idx = tid * 8, r = idx >> 5, kk = idx & 31;
    stage16(Q + (size_t)(qBase + r) * DK + k0 + kk, &Qs[r * LDK + kk]);
    stage16(K + (size_t)(sBase + r) * DK + k0 + kk, &Ks[r * LDK + kk]);
    stage_fence();
    __syncthreads();
    wmma_step(Qs, Ks, rB, cB, lane, acc0, acc1);
    __syncthreads();
  }

  const int n = cB + (lane & 15);
#pragma unroll
  for (int r8 = 0; r8 < 8; ++r8) {
    const int m = rB + r8 + ((lane & 16) >> 1);
    float* dst = attns + ((size_t)hb * SQ + qBase + m) * SQ + sBase;
    dst[n]      = acc0[r8] * INV_TEMPER;
    dst[n + 16] = acc1[r8] * INV_TEMPER;
  }
}

// ---------------- kernel 2b: row softmax in place ----------------
__global__ void __launch_bounds__(256)
softmax_rows(float* __restrict__ attns) {
  __shared__ float red[256];
  float* p = attns + (size_t)blockIdx.x * SQ;
  const int t = threadIdx.x;

  float v[8];
  float mx = -3.4e38f;
#pragma unroll
  for (int i = 0; i < 8; ++i) { v[i] = p[t + 256 * i]; mx = fmaxf(mx, v[i]); }
  red[t] = mx;
  for (int off = 128; off > 0; off >>= 1) {
    __syncthreads();
    if (t < off) red[t] = fmaxf(red[t], red[t + off]);
  }
  __syncthreads();
  const float m = red[0];
  float sum = 0.f;
#pragma unroll
  for (int i = 0; i < 8; ++i) { v[i] = __expf(v[i] - m); sum += v[i]; }
  __syncthreads();
  red[t] = sum;
  for (int off = 128; off > 0; off >>= 1) {
    __syncthreads();
    if (t < off) red[t] += red[t + off];
  }
  __syncthreads();
  const float inv = 1.0f / red[0];
#pragma unroll
  for (int i = 0; i < 8; ++i) p[t + 256 * i] = v[i] * inv;
}

// ---------------- kernel 3: O = P x V ----------------
// grid = (q-tile 32, hb 64); writes concat-head bf16 [8192][h*64+v]
__global__ void __launch_bounds__(256)
attn_pv(const float* __restrict__ attns,
        const unsigned short* __restrict__ vh,
        unsigned short* __restrict__ outh) {
  __shared__ __align__(16) unsigned short Ps[64 * LDK];
  __shared__ __align__(16) unsigned short Vt[64 * LDK];   // [v][s] transposed

  const int hb = blockIdx.y;
  const int h = hb >> 2, b = hb & 3;
  const float* P = attns + (size_t)hb * SQ * SQ;
  const unsigned short* V = vh + (size_t)hb * SQ * DK;
  const int qBase = blockIdx.x * 64;

  const int tid = threadIdx.x, lane = tid & 31, wv = tid >> 5;
  const int rB = (wv & 3) * 16, cB = (wv >> 2) * 32;

  v8f acc0 = {}; v8f acc1 = {};

  for (int s0 = 0; s0 < SQ; s0 += 32) {
    {  // P tile 64 q x 32 s, f32 -> bf16 into LDS (conversion needs VALU)
      const int idx = tid * 8, r = idx >> 5, ss = idx & 31;
      const float* src = P + (size_t)(qBase + r) * SQ + s0 + ss;
      float4 f0 = *(const float4*)(src);
      float4 f1 = *(const float4*)(src + 4);
      if (s0 + 32 < SQ) __builtin_prefetch(src + 32, 0, 1);
      uint4 pk;
      pk.x = (unsigned int)f2bf(f0.x) | ((unsigned int)f2bf(f0.y) << 16);
      pk.y = (unsigned int)f2bf(f0.z) | ((unsigned int)f2bf(f0.w) << 16);
      pk.z = (unsigned int)f2bf(f1.x) | ((unsigned int)f2bf(f1.y) << 16);
      pk.w = (unsigned int)f2bf(f1.z) | ((unsigned int)f2bf(f1.w) << 16);
      *(uint4*)&Ps[r * LDK + ss] = pk;
    }
    {  // V tile 32 s x 64 v, transpose into Vt[v][s]
      const int idx = tid * 8, ss = idx >> 6, v0 = idx & 63;
      uint4 d = *(const uint4*)(V + (size_t)(s0 + ss) * DK + v0);
      const unsigned short* u = (const unsigned short*)&d;
#pragma unroll
      for (int i = 0; i < 8; ++i) Vt[(v0 + i) * LDK + ss] = u[i];
    }
    __syncthreads();
    wmma_step(Ps, Vt, rB, cB, lane, acc0, acc1);
    __syncthreads();
  }

  const int n = cB + (lane & 15);
#pragma unroll
  for (int r8 = 0; r8 < 8; ++r8) {
    const int m = rB + r8 + ((lane & 16) >> 1);
    unsigned short* dst =
        outh + ((size_t)b * SQ + qBase + m) * DM + (size_t)h * DK;
    dst[n]      = f2bf(acc0[r8]);
    dst[n + 16] = f2bf(acc1[r8]);
  }
}

// ---------------- kernel 4: output projection + bias + residual ----------------
// y[row][m] = sum_j outh[row][j]*proj_w[m][j] + proj_b[m] + q[row][m]
__global__ void __launch_bounds__(256)
out_proj(const unsigned short* __restrict__ outh,
         const unsigned short* __restrict__ pw,
         const float* __restrict__ pb,
         const float* __restrict__ qin,
         float* __restrict__ y) {
  __shared__ __align__(16) unsigned short Asm[64 * LDK];
  __shared__ __align__(16) unsigned short Bsm[64 * LDK];   // [m][j]

  const int rowBase = blockIdx.x * 64;
  const int colBase = blockIdx.y * 64;
  const int tid = threadIdx.x, lane = tid & 31, wv = tid >> 5;
  const int rB = (wv & 3) * 16, cB = (wv >> 2) * 32;

  v8f acc0 = {}; v8f acc1 = {};

  for (int k0 = 0; k0 < DM; k0 += 32) {
    const int idx = tid * 8, r = idx >> 5, kk = idx & 31;
    const unsigned short* srcA = outh + (size_t)(rowBase + r) * DM + k0 + kk;
    const unsigned short* srcB = pw + (size_t)(colBase + r) * DM + k0 + kk;
    stage16(srcA, &Asm[r * LDK + kk]);
    stage16(srcB, &Bsm[r * LDK + kk]);   // proj_w [m][j]: j contiguous == K
    if (k0 + 32 < DM) {
      __builtin_prefetch(srcA + 32, 0, 1);
      __builtin_prefetch(srcB + 32, 0, 1);
    }
    stage_fence();
    __syncthreads();
    wmma_step(Asm, Bsm, rB, cB, lane, acc0, acc1);
    __syncthreads();
  }

  const int n = cB + (lane & 15);
#pragma unroll
  for (int r8 = 0; r8 < 8; ++r8) {
    const int m = rB + r8 + ((lane & 16) >> 1);
    const size_t row = (size_t)rowBase + m;
    const size_t c0 = (size_t)colBase + n;
    y[row * DM + c0]      = acc0[r8] + pb[c0]      + qin[row * DM + c0];
    y[row * DM + c0 + 16] = acc1[r8] + pb[c0 + 16] + qin[row * DM + c0 + 16];
  }
}

// ---------------- kernel 5: layernorm ----------------
__global__ void __launch_bounds__(256)
layernorm_rows(const float* __restrict__ y, const float* __restrict__ g,
               const float* __restrict__ be, float* __restrict__ out) {
  __shared__ float red[256];
  const size_t row = blockIdx.x;
  const float* x = y + row * DM;
  const int t = threadIdx.x;

  float v[4];
  float s = 0.f;
#pragma unroll
  for (int i = 0; i < 4; ++i) { v[i] = x[t + 256 * i]; s += v[i]; }
  red[t] = s;
  for (int off = 128; off > 0; off >>= 1) {
    __syncthreads();
    if (t < off) red[t] += red[t + off];
  }
  __syncthreads();
  const float mu = red[0] * (1.0f / DM);
  float vs = 0.f;
#pragma unroll
  for (int i = 0; i < 4; ++i) { float d = v[i] - mu; vs += d * d; }
  __syncthreads();
  red[t] = vs;
  for (int off = 128; off > 0; off >>= 1) {
    __syncthreads();
    if (t < off) red[t] += red[t + off];
  }
  __syncthreads();
  const float inv = rsqrtf(red[0] * (1.0f / DM) + LN_EPS);
  float* o = out + row * DM;
#pragma unroll
  for (int i = 0; i < 4; ++i) {
    const int c = t + 256 * i;
    o[c] = (v[i] - mu) * inv * g[c] + be[c];
  }
}

// ============================================================================
extern "C" void kernel_launch(void* const* d_in, const int* in_sizes, int n_in,
                              void* d_out, int out_size, void* d_ws, size_t ws_size,
                              hipStream_t stream) {
  (void)in_sizes; (void)n_in; (void)out_size; (void)ws_size;

  const float* q      = (const float*)d_in[0];
  const float* k      = (const float*)d_in[1];
  const float* v      = (const float*)d_in[2];
  const float* w_qs   = (const float*)d_in[3];
  const float* w_ks   = (const float*)d_in[4];
  const float* w_vs   = (const float*)d_in[5];
  const float* proj_w = (const float*)d_in[6];
  const float* proj_b = (const float*)d_in[7];
  const float* ln_g   = (const float*)d_in[8];
  const float* ln_b   = (const float*)d_in[9];

  char* ws = (char*)d_ws;
  float* out   = (float*)d_out;
  float* attns = out + (size_t)ROWS * DM;   // out first, attns second

  unsigned short* qbf = (unsigned short*)(ws + OFF_QBF);
  unsigned short* kbf = (unsigned short*)(ws + OFF_KBF);
  unsigned short* vbf = (unsigned short*)(ws + OFF_VBF);
  unsigned short* wqb = (unsigned short*)(ws + OFF_WQ);
  unsigned short* wkb = (unsigned short*)(ws + OFF_WK);
  unsigned short* wvb = (unsigned short*)(ws + OFF_WV);
  unsigned short* pwb = (unsigned short*)(ws + OFF_PW);
  unsigned short* qh  = (unsigned short*)(ws + OFF_QH);
  unsigned short* kh  = (unsigned short*)(ws + OFF_KH);
  unsigned short* vh  = (unsigned short*)(ws + OFF_VH);
  unsigned short* oh  = (unsigned short*)(ws + OFF_OH);
  float* ybuf = (float*)(ws + OFF_Y);

  const int nX = ROWS * DM;         // 8,388,608
  const int nW = NH * DM * DK;      // 1,048,576
  const int nP = DM * DM;           // 1,048,576
  dim3 blk(256);

  // 0) f32 -> bf16 conversions
  cvt_f32_bf16<<<(nX + 255) / 256, blk, 0, stream>>>(q, qbf, nX);
  cvt_f32_bf16<<<(nX + 255) / 256, blk, 0, stream>>>(k, kbf, nX);
  cvt_f32_bf16<<<(nX + 255) / 256, blk, 0, stream>>>(v, vbf, nX);
  cvt_f32_bf16<<<(nW + 255) / 256, blk, 0, stream>>>(w_qs, wqb, nW);
  cvt_f32_bf16<<<(nW + 255) / 256, blk, 0, stream>>>(w_ks, wkb, nW);
  cvt_f32_bf16<<<(nW + 255) / 256, blk, 0, stream>>>(w_vs, wvb, nW);
  cvt_f32_bf16<<<(nP + 255) / 256, blk, 0, stream>>>(proj_w, pwb, nP);

  // 1) per-head projections: qh/kh/vh  (48 GEMMs of [8192,1024]x[1024,64])
  gemm_head_proj<<<dim3(ROWS / 64, 1, 3 * NH), blk, 0, stream>>>(ws);

  // 2) scores + softmax (probs are an output, kept f32 in d_out)
  attn_scores<<<dim3(SQ / 64, SQ / 64, NH * BS), blk, 0, stream>>>(qh, kh, attns);
  softmax_rows<<<NH * BS * SQ, blk, 0, stream>>>(attns);

  // 3) O = P x V
  attn_pv<<<dim3(SQ / 64, NH * BS), blk, 0, stream>>>(attns, vh, oh);

  // 4) output projection + bias + residual
  out_proj<<<dim3(ROWS / 64, DM / 64), blk, 0, stream>>>(oh, pwb, proj_b, q, ybuf);

  // 5) layernorm -> d_out
  layernorm_rows<<<ROWS, blk, 0, stream>>>(ybuf, ln_g, ln_b, out);
}
